// UpdateRule_31198642438476
// MI455X (gfx1250) — compile-verified, matrix operand order checked by reading, not run on hardware
//
#include <hip/hip_runtime.h>
#include <hip/hip_bf16.h>

typedef __attribute__((ext_vector_type(16))) _Float16 v16h;
typedef __attribute__((ext_vector_type(8)))  float    v8f;

#define NN     100000
#define EE     1600000
#define NGRIDC 99872
#define HIDC   128
#define NP     100096   // ceil(NN/128)*128

// ---------- order-preserving float <-> u32 for atomic max ----------
__device__ __forceinline__ unsigned enc_f32(float f) {
    unsigned u = __float_as_uint(f);
    return (u & 0x80000000u) ? ~u : (u | 0x80000000u);
}
__device__ __forceinline__ float dec_f32(unsigned u) {
    unsigned b = (u & 0x80000000u) ? (u ^ 0x80000000u) : ~u;
    return __uint_as_float(b);
}
// fast sigmoid: v_exp_f32 + v_rcp_f32 (no IEEE divide chain)
__device__ __forceinline__ float sigmoidf(float x) {
    return __builtin_amdgcn_rcpf(1.0f + __expf(-x));
}

// ---------- prep: A f32[N][K] -> f16 [NP][KP] zero-padded ----------
__global__ void prep_a(const float* __restrict__ A, _Float16* __restrict__ O,
                       int K, int KP, int n) {
    int i = blockIdx.x * blockDim.x + threadIdx.x;
    if (i >= n) return;
    int row = i / KP, k = i - row * KP;
    float v = (row < NN && k < K) ? A[(size_t)row * K + k] : 0.0f;
    O[i] = (_Float16)v;
}

// ---------- prep: W f32[K][M] -> transposed f16 [MP][KP] zero-padded ----------
__global__ void prep_w(const float* __restrict__ W, _Float16* __restrict__ O,
                       int K, int M, int KP, int n) {
    int i = blockIdx.x * blockDim.x + threadIdx.x;
    if (i >= n) return;
    int m = i / KP, k = i - m * KP;
    float v = (m < M && k < K) ? W[(size_t)k * M + m] : 0.0f;
    O[i] = (_Float16)v;
}

// ---------- WMMA GEMM: out = act(Af16[NP,KP] @ Wt[MP,KP]^T + bias) ----------
// block = 256 threads = 8 wave32s; macro-tile 128x128; 8 v_wmma per wave per K-step.
// All epilogue configuration is compile-time -> branch-free tail.
template <bool HASBIAS, bool ACT, bool OUTF16, bool COLCHK>
__global__ __launch_bounds__(256) void gemm_wmma(
        const _Float16* __restrict__ Af16, const _Float16* __restrict__ Wt,
        const float* __restrict__ bias, float* __restrict__ outf32,
        _Float16* __restrict__ outf16, int Nrows, int KP, int M, int outKP) {
    __shared__ _Float16 As[128 * 40];   // [row][k], stride 40 halfs (16B aligned)
    __shared__ _Float16 Bs[128 * 40];   // [n][k]

    const int tid  = threadIdx.x;
    const int lane = tid & 31;
    const int wave = tid >> 5;
    const int row0 = blockIdx.x * 128;
    const int col0 = blockIdx.y * 128;
    const int nloc = lane & 15;

    // preload bias (1 coalesced load per sub-tile), issued before the K loop
    float bv[8];
#pragma unroll
    for (int s = 0; s < 8; ++s) {
        if (HASBIAS) {
            int gn = col0 + s * 16 + nloc;
            bv[s] = (!COLCHK || gn < M) ? bias[gn] : 0.0f;
        } else {
            bv[s] = 0.0f;
        }
    }

    v8f c[8];
#pragma unroll
    for (int s = 0; s < 8; ++s) c[s] = (v8f){};

    for (int kk = 0; kk < KP; kk += 32) {
        __syncthreads();
        // stage A tile 128x32 halfs: 512 x 16B chunks, branch-free
#pragma unroll
        for (int i = 0; i < 2; ++i) {
            int cidx = tid + i * 256;
            int r = cidx >> 2, co = (cidx & 3) * 8;
            *(uint4*)(As + r * 40 + co) =
                *(const uint4*)(Af16 + (size_t)(row0 + r) * KP + kk + co);
        }
#pragma unroll
        for (int i = 0; i < 2; ++i) {
            int cidx = tid + i * 256;
            int r = cidx >> 2, co = (cidx & 3) * 8;
            *(uint4*)(Bs + r * 40 + co) =
                *(const uint4*)(Wt + (size_t)(col0 + r) * KP + kk + co);
        }
        __syncthreads();

        // A fragment per ISA 16-bit layout: lane L -> M=L&15,
        // K pair base = (v>=4?16:0) + (L>=16?8:0) + 2*(v&3)
        v16h a;
        const int m  = wave * 16 + nloc;
        const int hi = (lane & 16) ? 8 : 0;
#pragma unroll
        for (int v = 0; v < 8; ++v) {
            int kb = ((v & 4) ? 16 : 0) + hi + ((v & 3) << 1);
            a[2 * v]     = As[m * 40 + kb];
            a[2 * v + 1] = As[m * 40 + kb + 1];
        }
#pragma unroll
        for (int s = 0; s < 8; ++s) {
            v16h b;
            const int n = s * 16 + nloc;
#pragma unroll
            for (int v = 0; v < 8; ++v) {
                int kb = ((v & 4) ? 16 : 0) + hi + ((v & 3) << 1);
                b[2 * v]     = Bs[n * 40 + kb];
                b[2 * v + 1] = Bs[n * 40 + kb + 1];
            }
            c[s] = __builtin_amdgcn_wmma_f32_16x16x32_f16(false, a, false, b,
                                                          (short)0, c[s], false, false);
        }
    }

    // D layout: vgpr j -> M = j + 8*(lane>=16), N = lane&15
    const int mb = wave * 16 + ((lane & 16) ? 8 : 0);
#pragma unroll
    for (int j = 0; j < 8; ++j) {
        const int gm = row0 + mb + j;
        if (OUTF16) {        // f16 path: M == outKP == 128, rows padded to NP
#pragma unroll
            for (int s = 0; s < 8; ++s) {
                int gn = col0 + s * 16 + nloc;
                float v = c[s][j] + bv[s];
                if (ACT) v = sigmoidf(v);
                outf16[(size_t)gm * outKP + gn] = (_Float16)v;
            }
        } else {
            if (gm < Nrows) {
#pragma unroll
                for (int s = 0; s < 8; ++s) {
                    int gn = col0 + s * 16 + nloc;
                    if (!COLCHK || gn < M) {
                        float v = c[s][j] + bv[s];
                        if (ACT) v = sigmoidf(v);
                        outf32[(size_t)gm * M + gn] = v;
                    }
                }
            }
        }
    }
}

// ---------- degree / normalization ----------
__global__ void zero_u32(unsigned* __restrict__ p, int n) {
    int i = blockIdx.x * blockDim.x + threadIdx.x;
    if (i < n) p[i] = 0u;
}
__global__ void zero_u32x4(uint4* __restrict__ p, int n4) {
    int i = blockIdx.x * blockDim.x + threadIdx.x;
    if (i < n4) p[i] = (uint4){0u, 0u, 0u, 0u};
}
__global__ void count_deg(const int* __restrict__ col, unsigned* __restrict__ cnt, int E) {
    int i = blockIdx.x * blockDim.x + threadIdx.x;
    if (i < E) atomicAdd(&cnt[col[i]], 1u);
}
__global__ void make_dinv(const unsigned* __restrict__ cnt, float* __restrict__ dinv, int n) {
    int i = blockIdx.x * blockDim.x + threadIdx.x;
    if (i < n) dinv[i] = rsqrtf((float)cnt[i] + 1.0f);   // +1 self-loop
}

// ---------- gate mix: xm = x*f + (1-f)*init_pad; also writes xm_f16 padded ----------
__global__ void mix_kernel(const float* __restrict__ x, const float* __restrict__ f,
                           const float* __restrict__ initial, float* __restrict__ xm,
                           _Float16* __restrict__ xmf16, int n /* NP*160 */) {
    int i = blockIdx.x * blockDim.x + threadIdx.x;
    if (i >= n) return;
    int row = i / 160, d = i - row * 160;
    float val = 0.0f;
    if (row < NN && d < 130) {
        float ip = 0.0f;
        if (d < HIDC && row >= NGRIDC) ip = initial[(size_t)(row - NGRIDC) * HIDC + d];
        float fv = f[(size_t)row * 130 + d];
        val = x[(size_t)row * 130 + d] * fv + (1.0f - fv) * ip;
        xm[(size_t)row * 130 + d] = val;
    }
    xmf16[i] = (_Float16)val;
}

// ---------- edge scatter-max: one wave32 per edge, float4 per lane ----------
__global__ void scatter_max(const float* __restrict__ hw,
                            const int* __restrict__ erow, const int* __restrict__ ecol,
                            const float* __restrict__ dinv, unsigned* __restrict__ agg,
                            int E, int total) {
    int e = blockIdx.x * 8 + (threadIdx.x >> 5);
    if (e >= total) return;
    int lane = threadIdx.x & 31;
    int r, c; float nrm;
    if (e < E) { r = erow[e]; c = ecol[e]; nrm = dinv[r] * dinv[c]; }
    else       { r = c = e - E; float d0 = dinv[r]; nrm = d0 * d0; }   // self-loop
    int d0 = lane * 4;
    float4 v = *(const float4*)(hw + (size_t)r * HIDC + d0);           // 512B/edge gather
    unsigned* ap = agg + (size_t)c * HIDC + d0;
    atomicMax(ap + 0, enc_f32(v.x * nrm));
    atomicMax(ap + 1, enc_f32(v.y * nrm));
    atomicMax(ap + 2, enc_f32(v.z * nrm));
    atomicMax(ap + 3, enc_f32(v.w * nrm));
}

// ---------- fused bias + LayerNorm (+ReLU) -> f16 A-operand for next GEMM ----------
__global__ void postln(const unsigned* __restrict__ agg, const float* __restrict__ b,
                       const float* __restrict__ g, const float* __restrict__ bn,
                       _Float16* __restrict__ outf16, int n, int dorelu) {
    int wave = threadIdx.x >> 5, lane = threadIdx.x & 31;
    int row = blockIdx.x * 8 + wave;
    if (row >= n) return;
    int d0 = lane * 4;
    uint4  av = *(const uint4*)(agg + (size_t)row * HIDC + d0);
    float4 bb = *(const float4*)(b + d0);
    float v[4];
    v[0] = dec_f32(av.x) + bb.x; v[1] = dec_f32(av.y) + bb.y;
    v[2] = dec_f32(av.z) + bb.z; v[3] = dec_f32(av.w) + bb.w;
    float s = v[0] + v[1] + v[2] + v[3];
#pragma unroll
    for (int o = 16; o >= 1; o >>= 1) s += __shfl_xor(s, o, 32);
    float mu = s * (1.0f / 128.0f);
    float q = 0.0f;
#pragma unroll
    for (int i = 0; i < 4; ++i) { float t = v[i] - mu; q += t * t; }
#pragma unroll
    for (int o = 16; o >= 1; o >>= 1) q += __shfl_xor(q, o, 32);
    float rs = rsqrtf(q * (1.0f / 128.0f) + 1e-5f);
    float4 gg = *(const float4*)(g + d0);
    float4 nb = *(const float4*)(bn + d0);
    float o0 = (v[0] - mu) * rs * gg.x + nb.x;
    float o1 = (v[1] - mu) * rs * gg.y + nb.y;
    float o2 = (v[2] - mu) * rs * gg.z + nb.z;
    float o3 = (v[3] - mu) * rs * gg.w + nb.w;
    if (dorelu) {
        o0 = fmaxf(o0, 0.0f); o1 = fmaxf(o1, 0.0f);
        o2 = fmaxf(o2, 0.0f); o3 = fmaxf(o3, 0.0f);
    }
    union { _Float16 h[4]; uint2 u; } pk;
    pk.h[0] = (_Float16)o0; pk.h[1] = (_Float16)o1;
    pk.h[2] = (_Float16)o2; pk.h[3] = (_Float16)o3;
    *(uint2*)(outf16 + (size_t)row * HIDC + d0) = pk.u;
}

// ---------- final: out = xm[:, :128] + (agg + bco) * u ----------
__global__ void final_kernel(const float* __restrict__ xm, const unsigned* __restrict__ agg,
                             const float* __restrict__ bco, const float* __restrict__ u,
                             float* __restrict__ out, int n) {
    int i = blockIdx.x * blockDim.x + threadIdx.x;
    if (i >= n) return;
    int row = i >> 7, d = i & 127;
    out[i] = xm[(size_t)row * 130 + d] + (dec_f32(agg[i]) + bco[d]) * u[i];
}

static inline int cdiv(int a, int b) { return (a + b - 1) / b; }

extern "C" void kernel_launch(void* const* d_in, const int* in_sizes, int n_in,
                              void* d_out, int out_size, void* d_ws, size_t ws_size,
                              hipStream_t stream) {
    const float* x       = (const float*)d_in[0];
    const int*   ei      = (const int*)d_in[1];
    const float* initial = (const float*)d_in[2];
    const float* Wf1 = (const float*)d_in[3];  const float* bf1 = (const float*)d_in[4];
    const float* Wf2 = (const float*)d_in[5];  const float* bf2 = (const float*)d_in[6];
    const float* Wu1 = (const float*)d_in[7];  const float* bu1 = (const float*)d_in[8];
    const float* Wu2 = (const float*)d_in[9];  const float* bu2 = (const float*)d_in[10];
    const float* Wc1 = (const float*)d_in[11]; const float* bc1 = (const float*)d_in[12];
    const float* Wc2 = (const float*)d_in[13]; const float* bc2 = (const float*)d_in[14];
    const float* Wc3 = (const float*)d_in[15]; const float* bc3 = (const float*)d_in[16];
    const float* Wc4 = (const float*)d_in[17]; const float* bc4 = (const float*)d_in[18];
    const float* Wco = (const float*)d_in[19]; const float* bco = (const float*)d_in[20];
    const float* g3  = (const float*)d_in[21]; const float* bn3 = (const float*)d_in[22];
    const float* g6  = (const float*)d_in[23]; const float* bn6 = (const float*)d_in[24];
    const float* g7  = (const float*)d_in[25]; const float* bn7 = (const float*)d_in[26];

    const int* erow = ei;
    const int* ecol = ei + EE;

    // ---- workspace layout: f32 region (all sizes multiple of 4 floats) ----
    float* ws = (float*)d_ws;
    size_t o = 0;
    float*    dinv = ws + o; o += (size_t)NN;
    float*    xm   = ws + o; o += (size_t)NN * 130;
    float*    u    = ws + o; o += (size_t)NN * 128;
    float*    hw   = ws + o; o += (size_t)NN * 128;
    float*    tmp  = ws + o; o += (size_t)NN * 130;     // f-gate [N,130]
    unsigned* agg  = (unsigned*)(ws + o); o += (size_t)NN * 128;
    // ---- f16 region (16B aligned by construction) ----
    _Float16* hp = (_Float16*)(ws + o);
    size_t ho = 0;
    _Float16* xf16  = hp + ho; ho += (size_t)NP * 160;
    _Float16* xmf16 = hp + ho; ho += (size_t)NP * 160;
    _Float16* tf16  = hp + ho; ho += (size_t)NP * 128;
    _Float16* hf16  = hp + ho; ho += (size_t)NP * 128;
    _Float16* wf1t  = hp + ho; ho += 128 * 160;   // [MP=128][KP=160]
    _Float16* wf2t  = hp + ho; ho += 256 * 128;   // [MP=256][KP=128]
    _Float16* wu1t  = hp + ho; ho += 128 * 160;
    _Float16* wu2t  = hp + ho; ho += 128 * 128;
    _Float16* wc1t  = hp + ho; ho += 128 * 160;
    _Float16* wc2t  = hp + ho; ho += 128 * 128;
    _Float16* wc3t  = hp + ho; ho += 128 * 128;
    _Float16* wc4t  = hp + ho; ho += 128 * 128;
    _Float16* wcot  = hp + ho; ho += 128 * 128;

    float* outp = (float*)d_out;
    const int total = EE + NN;

    // ---- operand prep: f16 conversion / transpose ----
    prep_a<<<cdiv(NP * 160, 256), 256, 0, stream>>>(x, xf16, 130, 160, NP * 160);
    prep_w<<<cdiv(128 * 160, 256), 256, 0, stream>>>(Wf1, wf1t, 130, 128, 160, 128 * 160);
    prep_w<<<cdiv(256 * 128, 256), 256, 0, stream>>>(Wf2, wf2t, 128, 130, 128, 256 * 128);
    prep_w<<<cdiv(128 * 160, 256), 256, 0, stream>>>(Wu1, wu1t, 130, 128, 160, 128 * 160);
    prep_w<<<cdiv(128 * 128, 256), 256, 0, stream>>>(Wu2, wu2t, 128, 128, 128, 128 * 128);
    prep_w<<<cdiv(128 * 160, 256), 256, 0, stream>>>(Wc1, wc1t, 130, 128, 160, 128 * 160);
    prep_w<<<cdiv(128 * 128, 256), 256, 0, stream>>>(Wc2, wc2t, 128, 128, 128, 128 * 128);
    prep_w<<<cdiv(128 * 128, 256), 256, 0, stream>>>(Wc3, wc3t, 128, 128, 128, 128 * 128);
    prep_w<<<cdiv(128 * 128, 256), 256, 0, stream>>>(Wc4, wc4t, 128, 128, 128, 128 * 128);
    prep_w<<<cdiv(128 * 128, 256), 256, 0, stream>>>(Wco, wcot, 128, 128, 128, 128 * 128);

    // ---- degree & symmetric normalization (agg reused as counter) ----
    zero_u32<<<cdiv(NN, 256), 256, 0, stream>>>(agg, NN);
    count_deg<<<cdiv(EE, 256), 256, 0, stream>>>(ecol, agg, EE);
    make_dinv<<<cdiv(NN, 256), 256, 0, stream>>>(agg, dinv, NN);

    const dim3 g1(NP / 128, 1);   // M=128 GEMMs
    const dim3 g2(NP / 128, 2);   // M=130 GEMM (MP=256)

    // ---- forget gate: f = sig(sig(x@Wf1+bf1)@Wf2+bf2); mix toward initial ----
    gemm_wmma<true, true, true, false><<<g1, 256, 0, stream>>>(
        xf16, wf1t, bf1, nullptr, tf16, NN, 160, 128, 128);
    gemm_wmma<true, true, false, true><<<g2, 256, 0, stream>>>(
        tf16, wf2t, bf2, tmp, nullptr, NN, 128, 130, 0);
    mix_kernel<<<cdiv(NP * 160, 256), 256, 0, stream>>>(x, tmp, initial, xm, xmf16, NP * 160);

    // ---- update gate u ----
    gemm_wmma<true, true, true, false><<<g1, 256, 0, stream>>>(
        xmf16, wu1t, bu1, nullptr, tf16, NN, 160, 128, 128);
    gemm_wmma<true, true, false, false><<<g1, 256, 0, stream>>>(
        tf16, wu2t, bu2, u, nullptr, NN, 128, 128, 0);

    const int sgrid = cdiv(total, 8);
    const int z4    = NN * 128 / 4;
    const int zgrid = cdiv(z4, 256);

    // ---- GCN layer 1: relu(ln(gcn(xm, Wc1) + bc1)) ----
    gemm_wmma<false, false, false, false><<<g1, 256, 0, stream>>>(
        xmf16, wc1t, nullptr, hw, nullptr, NN, 160, 128, 0);
    zero_u32x4<<<zgrid, 256, 0, stream>>>((uint4*)agg, z4);
    scatter_max<<<sgrid, 256, 0, stream>>>(hw, erow, ecol, dinv, agg, EE, total);
    postln<<<cdiv(NN, 8), 256, 0, stream>>>(agg, bc1, g3, bn3, hf16, NN, 1);

    // ---- GCN layer 2: ln(.., g3, bn3) ----
    gemm_wmma<false, false, false, false><<<g1, 256, 0, stream>>>(
        hf16, wc2t, nullptr, hw, nullptr, NN, 128, 128, 0);
    zero_u32x4<<<zgrid, 256, 0, stream>>>((uint4*)agg, z4);
    scatter_max<<<sgrid, 256, 0, stream>>>(hw, erow, ecol, dinv, agg, EE, total);
    postln<<<cdiv(NN, 8), 256, 0, stream>>>(agg, bc2, g3, bn3, hf16, NN, 0);

    // ---- GCN layer 3: ln(.., g6, bn6) ----
    gemm_wmma<false, false, false, false><<<g1, 256, 0, stream>>>(
        hf16, wc3t, nullptr, hw, nullptr, NN, 128, 128, 0);
    zero_u32x4<<<zgrid, 256, 0, stream>>>((uint4*)agg, z4);
    scatter_max<<<sgrid, 256, 0, stream>>>(hw, erow, ecol, dinv, agg, EE, total);
    postln<<<cdiv(NN, 8), 256, 0, stream>>>(agg, bc3, g6, bn6, hf16, NN, 0);

    // ---- GCN layer 4: ln(.., g7, bn7) ----
    gemm_wmma<false, false, false, false><<<g1, 256, 0, stream>>>(
        hf16, wc4t, nullptr, hw, nullptr, NN, 128, 128, 0);
    zero_u32x4<<<zgrid, 256, 0, stream>>>((uint4*)agg, z4);
    scatter_max<<<sgrid, 256, 0, stream>>>(hw, erow, ecol, dinv, agg, EE, total);
    postln<<<cdiv(NN, 8), 256, 0, stream>>>(agg, bc4, g7, bn7, hf16, NN, 0);

    // ---- GCN output layer + residual/update-gate combine ----
    gemm_wmma<false, false, false, false><<<g1, 256, 0, stream>>>(
        hf16, wcot, nullptr, hw, nullptr, NN, 128, 128, 0);
    zero_u32x4<<<zgrid, 256, 0, stream>>>((uint4*)agg, z4);
    scatter_max<<<sgrid, 256, 0, stream>>>(hw, erow, ecol, dinv, agg, EE, total);
    final_kernel<<<cdiv(NN * 128, 256), 256, 0, stream>>>(xm, agg, bco, u, outp, NN * 128);
}
